// DecoderRNN_26130581029506
// MI455X (gfx1250) — matrix operand order repaired and checked by reading, not compiled, not run on hardware
//
#include <hip/hip_runtime.h>
#include <hip/hip_bf16.h>
#include <stdint.h>

// Problem constants (match reference)
#define B_   32
#define T_   64
#define V_   30000
#define EMB_ 512
#define H_   512
#define E_   256
#define KX_  (EMB_ + E_)   // 768  (x = [emb_t, context])
#define KCAT (KX_ + H_)    // 1280 (x ++ h  ->  W_ih ++ W_hh)
#define G4_  (4 * H_)      // 2048 gate width

typedef __attribute__((ext_vector_type(16))) __bf16        v16bf;
typedef __attribute__((ext_vector_type(8)))  float          v8f;
typedef __attribute__((ext_vector_type(8)))  unsigned short ush8;

union FragU { ush8 u[2]; v16bf v; };

__device__ __forceinline__ unsigned short f2bf(float f) {
  union { float f; unsigned int u; } x; x.f = f;
  unsigned int u = x.u;
  unsigned int r = (u + 0x7fffu + ((u >> 16) & 1u)) >> 16;  // RNE
  return (unsigned short)r;
}

// ---- WMMA fragment loaders (wave32, bf16 16x16x32) ---------------------------
// A 16x32 (MxK), per ISA: lane L holds row M=L&15; halves j<8 -> K = base+8*(L>=16)+j,
// halves j>=8 -> K = base+8*(L>=16)+16+(j-8). Two contiguous 16B loads per lane.
__device__ __forceinline__ v16bf load_a(const unsigned short* src, int stride,
                                        int row, int kbase, int lane) {
  const unsigned short* p = src + (size_t)row * stride + kbase + ((lane >= 16) ? 8 : 0);
  FragU f;
  f.u[0] = *(const ush8*)(p);
  f.u[1] = *(const ush8*)(p + 16);
  return f.v;
}
// B 32x16 (KxN): lane L holds column N=L&15; halves j<8 -> K = base+8*(L>=16)+j, j>=8 -> K = base+8*(L>=16)+16+(j-8).
// Since gemm is X @ W^T, B[k][n] = W[n][k]: two 16B runs of row n of W, starting 32B apart.
__device__ __forceinline__ v16bf load_b(const unsigned short* w, int stride,
                                        int row, int kbase, int lane) {
  const unsigned short* p = w + (size_t)row * stride + kbase + ((lane >= 16) ? 8 : 0);
  FragU f;
  f.u[0] = *(const ush8*)(p);
  f.u[1] = *(const ush8*)(p + 16);
  return f.v;
}

__device__ __forceinline__ v8f wmma_bf16(v16bf a, v16bf b, v8f c) {
  return __builtin_amdgcn_wmma_f32_16x16x32_bf16(false, a, false, b,
                                                 (short)0, c, false, false);
}

// ---- One-time per-call prep --------------------------------------------------
__global__ void k_conv_wcat(const float* __restrict__ Wih, const float* __restrict__ Whh,
                            unsigned short* __restrict__ Wcat) {
  int i = blockIdx.x * blockDim.x + threadIdx.x;       // over 2048*1280
  if (i >= G4_ * KCAT) return;
  int n = i / KCAT, k = i % KCAT;
  float v = (k < KX_) ? Wih[(size_t)n * KX_ + k] : Whh[(size_t)n * H_ + (k - KX_)];
  Wcat[i] = f2bf(v);
}

__global__ void k_conv_wfc(const float* __restrict__ Wfc, unsigned short* __restrict__ out) {
  size_t i = (size_t)blockIdx.x * blockDim.x + threadIdx.x;
  if (i >= (size_t)V_ * H_) return;
  out[i] = f2bf(Wfc[i]);
}

// Xemb[t][b][0:512] = emb_table[captions[b,t]]; Xemb[t][b][512:768] = encoder_out[b]
// (softmax over a length-1 axis == 1, so context == encoder_out exactly)
__global__ void k_build_xemb(const int* __restrict__ captions, const float* __restrict__ emb_table,
                             const float* __restrict__ enc, unsigned short* __restrict__ Xemb) {
  int i = blockIdx.x * blockDim.x + threadIdx.x;       // over T*B*KX
  if (i >= T_ * B_ * KX_) return;
  int k = i % KX_; int tb = i / KX_; int b = tb % B_; int t = tb / B_;
  float v;
  if (k < EMB_) { int cap = captions[b * T_ + t]; v = emb_table[(size_t)cap * EMB_ + k]; }
  else          { v = enc[b * E_ + (k - EMB_)]; }
  Xemb[i] = f2bf(v);
}

__global__ void k_init_state(float* __restrict__ c, unsigned short* __restrict__ hbf) {
  int i = blockIdx.x * blockDim.x + threadIdx.x;
  if (i < B_ * H_) { c[i] = 0.0f; hbf[i] = 0; }
}

// ---- Per-step: gates GEMM  [32 x 2048] = x(1280) @ Wcat^T + bias -------------
// Block = 128 threads (4 waves). Wave w: M-tile (w&1)*16, N-tile blockIdx.x*32 + (w>>1)*16.
// K loop split at the Xemb/h region boundary (768, a multiple of 32) so each loop
// has a single base pointer -> no per-iteration address selects.
__global__ void __launch_bounds__(128) k_gates(const unsigned short* __restrict__ Xemb_t,
                                               const unsigned short* __restrict__ hbf,
                                               const unsigned short* __restrict__ Wcat,
                                               const float* __restrict__ bih,
                                               const float* __restrict__ bhh,
                                               float* __restrict__ gates) {
  const int lane = threadIdx.x & 31, w = threadIdx.x >> 5;
  const int m0 = (w & 1) * 16;
  const int n0 = blockIdx.x * 32 + (w >> 1) * 16;
  const int aRow = m0 + (lane & 15);
  const int bRow = n0 + (lane & 15);          // output column == row of Wcat
  const float bias = bih[bRow] + bhh[bRow];
  v8f acc;
#pragma unroll
  for (int r = 0; r < 8; ++r) acc[r] = bias;

#pragma unroll 4
  for (int kk = 0; kk < KX_; kk += 32) {      // x part: K = 0..767 from Xemb[t]
    v16bf a = load_a(Xemb_t, KX_, aRow, kk, lane);
    v16bf b = load_b(Wcat, KCAT, bRow, kk, lane);
    acc = wmma_bf16(a, b, acc);
  }
#pragma unroll 4
  for (int kk = 0; kk < H_; kk += 32) {       // h part: K = 768..1279 from h
    v16bf a = load_a(hbf, H_, aRow, kk, lane);
    v16bf b = load_b(Wcat, KCAT, bRow, KX_ + kk, lane);
    acc = wmma_bf16(a, b, acc);
  }

  const int mb = m0 + ((lane >= 16) ? 8 : 0);
  const int col = n0 + (lane & 15);
#pragma unroll
  for (int r = 0; r < 8; ++r)
    gates[(size_t)(mb + r) * G4_ + col] = acc[r];
}

// ---- Per-step: LSTM elementwise (fp32 state, bf16 h for the GEMMs) -----------
__global__ void k_lstm_elem(const float* __restrict__ gates, float* __restrict__ c,
                            unsigned short* __restrict__ hbf) {
  int i = blockIdx.x * blockDim.x + threadIdx.x;
  if (i >= B_ * H_) return;
  int b = i / H_, j = i % H_;
  const float* g = gates + (size_t)b * G4_;
  float ig = g[j], fg = g[j + H_], gg = g[j + 2 * H_], og = g[j + 3 * H_];
  float si = 1.0f / (1.0f + __expf(-ig));
  float sf = 1.0f / (1.0f + __expf(-fg));
  float so = 1.0f / (1.0f + __expf(-og));
  float cn = sf * c[i] + si * tanhf(gg);
  c[i]   = cn;
  hbf[i] = f2bf(so * tanhf(cn));
}

// ---- Per-step: FC GEMM  out_t[32 x 30000] = h(512) @ Wfc^T + b_fc ------------
// Block = 128 threads (4 waves). Wave w: M-tile (w&1)*16, four N-tiles at
// blockIdx.x*128 + (w>>1)*64 (+0,+16,+32,+48): one A fragment feeds 4 WMMAs.
__global__ void __launch_bounds__(128) k_fc(const unsigned short* __restrict__ hbf,
                                            const unsigned short* __restrict__ Wfc,
                                            const float* __restrict__ bfc,
                                            float* __restrict__ out_t) {
  const int lane = threadIdx.x & 31, w = threadIdx.x >> 5;
  const int m0 = (w & 1) * 16;
  const int nb = blockIdx.x * 128 + (w >> 1) * 64;
  const int ln = lane & 15;
  int rrow[4];
  v8f acc[4];
#pragma unroll
  for (int q = 0; q < 4; ++q) {
    rrow[q] = min(nb + q * 16 + ln, V_ - 1);       // clamped Wfc row (tail guard)
    const float bq = bfc[rrow[q]];
#pragma unroll
    for (int r = 0; r < 8; ++r) acc[q][r] = bq;
  }
  const int aRow = m0 + ln;

#pragma unroll 2
  for (int kk = 0; kk < H_; kk += 32) {
    v16bf a = load_a(hbf, H_, aRow, kk, lane);
#pragma unroll
    for (int q = 0; q < 4; ++q) {
      v16bf b = load_b(Wfc, H_, rrow[q], kk, lane);
      acc[q] = wmma_bf16(a, b, acc[q]);
    }
  }

  const int mb = m0 + ((lane >= 16) ? 8 : 0);
  const size_t rs = (size_t)T_ * V_;               // out is [B, T, V]; out_t = d_out + t*V
#pragma unroll
  for (int q = 0; q < 4; ++q) {
    const int col = nb + q * 16 + ln;
    if (col < V_) {
#pragma unroll
      for (int r = 0; r < 8; ++r)
        out_t[(size_t)(mb + r) * rs + col] = acc[q][r];
    }
  }
}

// -----------------------------------------------------------------------------
extern "C" void kernel_launch(void* const* d_in, const int* in_sizes, int n_in,
                              void* d_out, int out_size, void* d_ws, size_t ws_size,
                              hipStream_t stream) {
  (void)in_sizes; (void)n_in; (void)out_size; (void)ws_size;
  const float* enc  = (const float*)d_in[0];   // [32,256]
  const int*   caps = (const int*)d_in[1];     // [32,64]
  // d_in[2..7] = We,be,Wd,bd,Wf,bf : attention over a length-1 sequence is
  // identically alpha==1 -> context == encoder_out exactly; params are inert.
  const float* emb  = (const float*)d_in[8];   // [30000,512]
  const float* Wih  = (const float*)d_in[9];   // [2048,768]
  const float* Whh  = (const float*)d_in[10];  // [2048,512]
  const float* bih  = (const float*)d_in[11];  // [2048]
  const float* bhh  = (const float*)d_in[12];  // [2048]
  const float* Wfc  = (const float*)d_in[13];  // [30000,512]
  const float* bfc  = (const float*)d_in[14];  // [30000]
  float* out = (float*)d_out;                  // [32,64,30000]

  char* ws = (char*)d_ws;
  auto alloc = [&](size_t bytes) {
    char* p = ws; ws += (bytes + 255) & ~(size_t)255; return p;
  };
  unsigned short* Wcat   = (unsigned short*)alloc((size_t)G4_ * KCAT * 2);   // 5.25 MB
  unsigned short* WfcB   = (unsigned short*)alloc((size_t)V_ * H_ * 2);      // 30.7 MB
  unsigned short* Xemb   = (unsigned short*)alloc((size_t)T_ * B_ * KX_ * 2);// 3.15 MB
  float*          gates  = (float*)alloc((size_t)B_ * G4_ * 4);              // 256 KB
  float*          cstate = (float*)alloc((size_t)B_ * H_ * 4);               // 64 KB
  unsigned short* hbf    = (unsigned short*)alloc((size_t)B_ * H_ * 2);      // 32 KB

  // One-time (per call) prep: bf16 weight copies, embedding gather, zero state.
  k_conv_wcat<<<(G4_ * KCAT + 255) / 256, 256, 0, stream>>>(Wih, Whh, Wcat);
  k_conv_wfc<<<(int)(((size_t)V_ * H_ + 255) / 256), 256, 0, stream>>>(Wfc, WfcB);
  k_build_xemb<<<(T_ * B_ * KX_ + 255) / 256, 256, 0, stream>>>(caps, emb, enc, Xemb);
  k_init_state<<<(B_ * H_ + 255) / 256, 256, 0, stream>>>(cstate, hbf);

  // Sequential recurrence: 64 steps x {gates GEMM, LSTM elementwise, FC GEMM}.
  for (int t = 0; t < T_; ++t) {
    const unsigned short* Xemb_t = Xemb + (size_t)t * B_ * KX_;
    k_gates<<<G4_ / 32, 128, 0, stream>>>(Xemb_t, hbf, Wcat, bih, bhh, gates);
    k_lstm_elem<<<(B_ * H_ + 255) / 256, 256, 0, stream>>>(gates, cstate, hbf);
    k_fc<<<(V_ + 127) / 128, 128, 0, stream>>>(hbf, WfcB, bfc, out + (size_t)t * V_);
  }
}
